// TopKRouter_24653112279327
// MI455X (gfx1250) — matrix operand order repaired and checked by reading, not compiled, not run on hardware
//
#include <hip/hip_runtime.h>

typedef __attribute__((ext_vector_type(16))) __bf16   v16bf;
typedef __attribute__((ext_vector_type(8)))  float    v8f;
typedef __attribute__((ext_vector_type(4)))  float    f32x4;
typedef __attribute__((ext_vector_type(4)))  unsigned u32x4;
typedef __attribute__((ext_vector_type(2)))  unsigned u32x2;
typedef int v4i __attribute__((vector_size(16)));   // matches builtin's pointee type

// Problem dims (match the reference)
constexpr int kTokens = 16384;   // B*S
constexpr int kDim    = 2048;    // D
constexpr int kExp    = 64;      // E
constexpr int kTop    = 2;       // K
constexpr int kRows   = 64;      // token rows per block (4 stripes x 16)
constexpr int kKC     = 32;      // K-chunk = one WMMA K-step
constexpr int kIters  = kDim / kKC;
constexpr int kWRow   = 20;      // padded bf16 W row stride in dwords (40 bf16 = 80B)
constexpr int kLPad   = kExp + 1;

__device__ __forceinline__ unsigned f2u(float f) { return __builtin_bit_cast(unsigned, f); }
__device__ __forceinline__ float    u2f(unsigned u) { return __builtin_bit_cast(float, u); }

// pack bf16(a),bf16(b) (truncated) into one dword via v_perm_b32
__device__ __forceinline__ unsigned phi(float a, float b) {
  return __builtin_amdgcn_perm(f2u(b), f2u(a), 0x07060302u);
}
// pack bf16 of the residuals (a - bf16(a)), (b - bf16(b))
__device__ __forceinline__ unsigned plo(float a, float b) {
  float ah = u2f(f2u(a) & 0xffff0000u);
  float bh = u2f(f2u(b) & 0xffff0000u);
  return phi(a - ah, b - bh);
}

__device__ __forceinline__ v8f wmma_bf16(v16bf a, v16bf b, v8f c) {
  return __builtin_amdgcn_wmma_f32_16x16x32_bf16(false, a, false, b,
                                                 (short)0, c, false, false);
}

#if __has_builtin(__builtin_amdgcn_global_load_async_to_lds_b128)
#define HAS_ASYNC_LDS 1
__device__ __forceinline__ void async_b128(const void* g, void* l) {
  __builtin_amdgcn_global_load_async_to_lds_b128(
      (__attribute__((address_space(1))) v4i*)g,
      (__attribute__((address_space(3))) v4i*)l, 0, 0);
}
__device__ __forceinline__ void wait_async0() {
#if __has_builtin(__builtin_amdgcn_s_wait_asynccnt)
  __builtin_amdgcn_s_wait_asynccnt(0);
#else
  asm volatile("s_wait_asynccnt 0" ::: "memory");
#endif
}
#else
#define HAS_ASYNC_LDS 0
#endif

__global__ __launch_bounds__(256)
void topk_router_wmma(const float* __restrict__ x,      // [kTokens, kDim]
                      const float* __restrict__ w,      // [kExp, kDim]
                      float* __restrict__ out_idx,      // [kTokens*kTop] (indices as float)
                      float* __restrict__ out_tp,       // [kTokens*kTop]
                      float* __restrict__ out_probs)    // [kTokens, kExp]
{
  __shared__ float    lds_x[2][kRows * kKC];        // 2 x 8 KB fp32 x tile (double buffer)
  __shared__ unsigned lds_wh[2][kExp * kWRow];      // 2 x 5 KB bf16-hi W tile
  __shared__ unsigned lds_wl[2][kExp * kWRow];      // 2 x 5 KB bf16-lo W tile
  __shared__ float    lds_logits[kRows * kLPad];    // ~16.6 KB

  const int tid    = threadIdx.x;
  const int lane   = tid & 31;
  const int wv     = tid >> 5;     // 0..7
  const int l16    = lane & 15;
  const int hs     = lane >> 4;    // half-wave select
  const int stripe = wv >> 1;      // 0..3 -> 16-row stripe
  const int th     = wv & 1;       // expert half: tiles {2*th, 2*th+1}
  const int rowBase = blockIdx.x * kRows;

  // ---- per-thread staging descriptors (advanced additively, no per-iter remul) ----
  const int f4a = tid,  ra = f4a >> 3, ca = f4a & 7;   // 512 f32x4 per tile / 256 thr
  const int f4b = 256 + tid, rb = f4b >> 3, cb = f4b & 7;
  const float* xga = x + (size_t)(rowBase + ra) * kDim + ca * 4;
  const float* xgb = x + (size_t)(rowBase + rb) * kDim + cb * 4;
  const float* wga = w + (size_t)ra * kDim + ca * 4;   // same decomposition: 64 rows x 8 f4
  const float* wgb = w + (size_t)rb * kDim + cb * 4;
  const int wda = ra * kWRow + ca * 2;                 // bf16 W store dword offsets
  const int wdb = rb * kWRow + cb * 2;

  v8f acc[2];
  #pragma unroll
  for (int t = 0; t < 2; ++t)
    #pragma unroll
    for (int i = 0; i < 8; ++i) acc[t][i] = 0.0f;

  // ---- prologue: start chunk 0 ----
#if HAS_ASYNC_LDS
  async_b128(xga, &lds_x[0][f4a * 4]);
  async_b128(xgb, &lds_x[0][f4b * 4]);
#else
  f32x4 xra = *(const f32x4*)xga;
  f32x4 xrb = *(const f32x4*)xgb;
#endif
  f32x4 wra = *(const f32x4*)wga;
  f32x4 wrb = *(const f32x4*)wgb;

  for (int i = 0; i < kIters; ++i) {
    const int cur = i & 1;
    const int nxt = cur ^ 1;

#if HAS_ASYNC_LDS
    wait_async0();                         // chunk i landed in lds_x[cur]
#else
    *(f32x4*)&lds_x[cur][f4a * 4] = xra;   // register-pipelined fallback
    *(f32x4*)&lds_x[cur][f4b * 4] = xrb;
#endif
    // convert + store W chunk i (regs were loaded last iteration)
    {
      u32x2 hv = { phi(wra.x, wra.y), phi(wra.z, wra.w) };
      u32x2 lv = { plo(wra.x, wra.y), plo(wra.z, wra.w) };
      *(u32x2*)&lds_wh[cur][wda] = hv;
      *(u32x2*)&lds_wl[cur][wda] = lv;
      hv = u32x2{ phi(wrb.x, wrb.y), phi(wrb.z, wrb.w) };
      lv = u32x2{ plo(wrb.x, wrb.y), plo(wrb.z, wrb.w) };
      *(u32x2*)&lds_wh[cur][wdb] = hv;
      *(u32x2*)&lds_wl[cur][wdb] = lv;
    }
    __syncthreads();                       // chunk i fully visible; nxt buffers free

    // ---- kick off chunk i+1 while we compute chunk i ----
    if (i + 1 < kIters) {
      xga += kKC; xgb += kKC; wga += kKC; wgb += kKC;
#if HAS_ASYNC_LDS
      async_b128(xga, &lds_x[nxt][f4a * 4]);
      async_b128(xgb, &lds_x[nxt][f4b * 4]);
#else
      xra = *(const f32x4*)xga;
      xrb = *(const f32x4*)xgb;
#endif
      wra = *(const f32x4*)wga;
      wrb = *(const f32x4*)wgb;
    }

    // ---- A fragments (16x32 bf16 layout): two runs of 8 consecutive k per lane ----
    union AF { v16bf bf; unsigned u[8]; } ah, al;
    {
      const f32x4* xr = (const f32x4*)&lds_x[cur][(stripe * 16 + l16) * kKC + hs * 8];
      f32x4 f0 = xr[0], f1 = xr[1];   // k = hs*8 .. hs*8+7
      f32x4 f2 = xr[4], f3 = xr[5];   // k = 16+hs*8 .. 16+hs*8+7
      ah.u[0] = phi(f0.x, f0.y);  al.u[0] = plo(f0.x, f0.y);
      ah.u[1] = phi(f0.z, f0.w);  al.u[1] = plo(f0.z, f0.w);
      ah.u[2] = phi(f1.x, f1.y);  al.u[2] = plo(f1.x, f1.y);
      ah.u[3] = phi(f1.z, f1.w);  al.u[3] = plo(f1.z, f1.w);
      ah.u[4] = phi(f2.x, f2.y);  al.u[4] = plo(f2.x, f2.y);
      ah.u[5] = phi(f2.z, f2.w);  al.u[5] = plo(f2.z, f2.w);
      ah.u[6] = phi(f3.x, f3.y);  al.u[6] = plo(f3.x, f3.y);
      ah.u[7] = phi(f3.z, f3.w);  al.u[7] = plo(f3.z, f3.w);
    }

    // ---- B fragments: pure ds_load_b128 of pre-converted bf16, 3-term WMMA ----
    #pragma unroll
    for (int t = 0; t < 2; ++t) {
      int tg = th * 2 + t;                  // global expert tile 0..3
      union BF { v16bf bf; u32x4 q[2]; } bh, bl;
      const u32x4* ph = (const u32x4*)&lds_wh[cur][(tg * 16 + l16) * kWRow + hs * 4];
      const u32x4* pl = (const u32x4*)&lds_wl[cur][(tg * 16 + l16) * kWRow + hs * 4];
      bh.q[0] = ph[0]; bh.q[1] = ph[2];
      bl.q[0] = pl[0]; bl.q[1] = pl[2];
      // hi*hi + hi*lo + lo*hi ~= fp32 product, f32 accumulation
      acc[t] = wmma_bf16(ah.bf, bh.bf, acc[t]);
      acc[t] = wmma_bf16(ah.bf, bl.bf, acc[t]);
      acc[t] = wmma_bf16(al.bf, bh.bf, acc[t]);
    }
    __syncthreads();
  }

  // ---- spill logits (C layout: VGPR r -> row r + 8*hs, col = 16*tg + l16) ----
  #pragma unroll
  for (int t = 0; t < 2; ++t) {
    int tg = th * 2 + t;
    #pragma unroll
    for (int r = 0; r < 8; ++r) {
      int row = stripe * 16 + r + hs * 8;
      int col = tg * 16 + l16;
      lds_logits[row * kLPad + col] = acc[t][r];
    }
  }
  __syncthreads();

  // ---- softmax + top-2 + renorm: one thread per row ----
  if (tid < kRows) {
    const float* lrow = &lds_logits[tid * kLPad];
    float m = lrow[0];
    #pragma unroll 8
    for (int e = 1; e < kExp; ++e) m = fmaxf(m, lrow[e]);

    float sum = 0.f;
    float b0 = -1.f, b1 = -1.f;
    int   i0 = 0,    i1 = 0;
    for (int e = 0; e < kExp; ++e) {
      float p = __expf(lrow[e] - m);
      sum += p;
      if (p > b0)      { b1 = b0; i1 = i0; b0 = p; i0 = e; }
      else if (p > b1) { b1 = p;  i1 = e; }
    }
    float inv = 1.0f / sum;
    size_t grow = (size_t)rowBase + tid;

    float* pr = out_probs + grow * kExp;
    #pragma unroll 8
    for (int e = 0; e < kExp; ++e) pr[e] = __expf(lrow[e] - m) * inv;

    float p0 = b0 * inv, p1 = b1 * inv;
    float s  = p0 + p1 + 1e-9f;
    out_idx[grow * 2 + 0] = (float)i0;
    out_idx[grow * 2 + 1] = (float)i1;
    out_tp [grow * 2 + 0] = p0 / s;
    out_tp [grow * 2 + 1] = p1 / s;
  }
}

extern "C" void kernel_launch(void* const* d_in, const int* in_sizes, int n_in,
                              void* d_out, int out_size, void* d_ws, size_t ws_size,
                              hipStream_t stream) {
  (void)in_sizes; (void)n_in; (void)out_size; (void)d_ws; (void)ws_size;
  const float* x = (const float*)d_in[0];
  const float* w = (const float*)d_in[1];
  float* out       = (float*)d_out;
  float* out_idx   = out;                                  // [16384*2]
  float* out_tp    = out + (size_t)kTokens * kTop;         // [16384*2]
  float* out_probs = out + (size_t)kTokens * kTop * 2;     // [16384*64]

  dim3 grid(kTokens / kRows);   // 256 blocks
  dim3 block(256);              // 8 waves
  hipLaunchKernelGGL(topk_router_wmma, grid, block, 0, stream,
                     x, w, out_idx, out_tp, out_probs);
}
